// GroupLinear_50835232915532
// MI455X (gfx1250) — matrix operand order repaired and checked by reading, not compiled, not run on hardware
//
#include <hip/hip_runtime.h>

// ---------------------------------------------------------------------------
// Grouped Linear (out = x @ w^T), fp32 in/out, bf16x3 WMMA emulation on
// gfx1250: x = xh + xl, w = wh + wl; out ~= xh*wh + xh*wl + xl*wh with fp32
// accumulation via v_wmma_f32_16x16x32_bf16.  Compute-bound (AI ~296 FLOP/B).
// Software-pipelined global loads + double-buffered LDS (80 KB of the 320 KB
// WGP pool), truncation-hi / RNE-lo split with packed 64-bit LDS stores.
// ---------------------------------------------------------------------------

typedef __attribute__((ext_vector_type(16))) __bf16        v16bf;
typedef __attribute__((ext_vector_type(8)))  __bf16        v8bf;
typedef __attribute__((ext_vector_type(8)))  float         v8f;
typedef __attribute__((ext_vector_type(4)))  float         v4f;
typedef __attribute__((ext_vector_type(2)))  unsigned int  v2u;

#define BM 128
#define BN 128
#define BK 32
#define STRIDE 40                 // 32 + 8 pad (80B rows): conflict-free b128
#define BUFHALF (4 * BM * STRIDE) // ushorts per LDS stage (Ah,Al,Bh,Bl)

// Split f into bf16 hi (truncation: the masked fp32 IS the hi value, so the
// residual is exact) and bf16 lo (RNE).  Packs two elements per dword.
__device__ __forceinline__ void split_pair(float f0, float f1,
                                           unsigned& hp, unsigned& lp) {
    unsigned u0 = __float_as_uint(f0);
    unsigned u1 = __float_as_uint(f1);
    unsigned h0 = u0 & 0xFFFF0000u;
    unsigned h1 = u1 & 0xFFFF0000u;
    float l0f = f0 - __uint_as_float(h0);
    float l1f = f1 - __uint_as_float(h1);
    unsigned l0 = __float_as_uint(l0f); l0 += 0x7FFFu + ((l0 >> 16) & 1u);
    unsigned l1 = __float_as_uint(l1f); l1 += 0x7FFFu + ((l1 >> 16) & 1u);
    hp = (u0 >> 16) | h1;                       // bf16 pair [f1h : f0h]
    lp = (l0 >> 16) | (l1 & 0xFFFF0000u);       // bf16 pair [f1l : f0l]
}

// Split a float4 into packed-bf16 hi/lo dword pairs.
__device__ __forceinline__ void split_quad(v4f f, v2u& hv, v2u& lv) {
    unsigned h0, l0, h1, l1;
    split_pair(f[0], f[1], h0, l0);
    split_pair(f[2], f[3], h1, l1);
    v2u h; h[0] = h0; h[1] = h1;
    v2u l; l[0] = l0; l[1] = l1;
    hv = h; lv = l;
}

// A fragment (16x32 bf16): lane&15 = M row; lanes 0-15 hold K {0..7,16..23},
// lanes 16-31 hold K {8..15,24..31}  (ISA 05_wmma 16-bit A layout).
__device__ __forceinline__ v16bf frag_a(const unsigned short* s, int lane) {
    const int m  = lane & 15;
    const int k0 = (lane & 16) ? 8 : 0;
    v8bf lo = *(const v8bf*)(s + m * STRIDE + k0);
    v8bf hi = *(const v8bf*)(s + m * STRIDE + k0 + 16);
    return __builtin_shufflevector(lo, hi, 0,1,2,3,4,5,6,7,8,9,10,11,12,13,14,15);
}

// B fragment (32x16 bf16): lane&15 = N col; lanes 0-15 hold K {0..7,16..23},
// lanes 16-31 hold K {8..15,24..31} (same K order as frag_a).
__device__ __forceinline__ v16bf frag_b(const unsigned short* s, int lane) {
    const int n  = lane & 15;
    const int k0 = (lane & 16) ? 8 : 0;
    v8bf lo = *(const v8bf*)(s + n * STRIDE + k0);
    v8bf hi = *(const v8bf*)(s + n * STRIDE + k0 + 16);
    return __builtin_shufflevector(lo, hi, 0,1,2,3,4,5,6,7,8,9,10,11,12,13,14,15);
}

struct LoadRegs { v4f a[4]; v4f b[4]; };   // 32 VGPRs of in-flight tile data

__device__ __forceinline__ void issue_loads(const float* __restrict__ X,
                                            const float* __restrict__ W,
                                            int mBlock, int nBlock, int K,
                                            int kb, int ldRow, int ldC,
                                            LoadRegs& r) {
#pragma unroll
    for (int p = 0; p < 4; ++p) {
        const int row = p * 32 + ldRow;
        r.a[p] = *(const v4f*)(X + (size_t)(mBlock + row) * K + kb + ldC);
        r.b[p] = *(const v4f*)(W + (size_t)(nBlock + row) * K + kb + ldC);
    }
}

__device__ __forceinline__ void convert_store(const LoadRegs& r,
                                              unsigned short* buf,
                                              int ldRow, int ldC) {
    unsigned short* sAh = buf;
    unsigned short* sAl = buf + 1 * BM * STRIDE;
    unsigned short* sBh = buf + 2 * BM * STRIDE;
    unsigned short* sBl = buf + 3 * BM * STRIDE;
#pragma unroll
    for (int p = 0; p < 4; ++p) {
        const int row = p * 32 + ldRow;
        v2u ah, al, bh, bl;
        split_quad(r.a[p], ah, al);
        split_quad(r.b[p], bh, bl);
        *(v2u*)(sAh + row * STRIDE + ldC) = ah;   // 8B packed bf16 stores
        *(v2u*)(sAl + row * STRIDE + ldC) = al;
        *(v2u*)(sBh + row * STRIDE + ldC) = bh;
        *(v2u*)(sBl + row * STRIDE + ldC) = bl;
    }
}

__global__ __launch_bounds__(256)
void group_gemm_bf16x3(const float* __restrict__ X,   // [M,K]
                       const float* __restrict__ W,   // [N,K]
                       float* __restrict__ Out,       // [M,N]
                       int M, int N, int K)
{
    __shared__ __align__(16) unsigned short smem[2 * BUFHALF];  // 80 KB

    const int tid  = threadIdx.x;
    const int lane = tid & 31;
    const int wave = tid >> 5;       // 8 waves
    const int wm   = wave & 1;       // 2 wave-rows  -> 64 M each
    const int wn   = wave >> 1;      // 4 wave-cols  -> 32 N each

    const int mBlock = blockIdx.y * BM;
    const int nBlock = blockIdx.x * BN;

    const int ldRow = tid >> 3;        // 0..31
    const int ldC   = (tid & 7) << 2;  // 0,4,...,28

    v8f acc[4][2] = {};

    // Prologue: stage tile 0 into LDS buffer 0.
    LoadRegs cur;
    issue_loads(X, W, mBlock, nBlock, K, 0, ldRow, ldC, cur);
    convert_store(cur, smem, ldRow, ldC);

    int pbuf = 0;
    for (int kb = 0; kb < K; kb += BK) {
        __syncthreads();                       // stage pbuf visible to all
        const bool more = (kb + BK) < K;

        LoadRegs nxt;
        if (more)                              // HBM loads fly over the WMMAs
            issue_loads(X, W, mBlock, nBlock, K, kb + BK, ldRow, ldC, nxt);

        unsigned short* buf = smem + pbuf * BUFHALF;
        unsigned short* sAh = buf;
        unsigned short* sAl = buf + 1 * BM * STRIDE;
        unsigned short* sBh = buf + 2 * BM * STRIDE;
        unsigned short* sBl = buf + 3 * BM * STRIDE;

        v16bf Ah[4], Al[4];
#pragma unroll
        for (int mt = 0; mt < 4; ++mt) {
            const int off = (wm * 64 + mt * 16) * STRIDE;
            Ah[mt] = frag_a(sAh + off, lane);
            Al[mt] = frag_a(sAl + off, lane);
        }
        v16bf Bh[2], Bl[2];
#pragma unroll
        for (int nt = 0; nt < 2; ++nt) {
            const int off = (wn * 32 + nt * 16) * STRIDE;
            Bh[nt] = frag_b(sBh + off, lane);
            Bl[nt] = frag_b(sBl + off, lane);
        }

#pragma unroll
        for (int mt = 0; mt < 4; ++mt)
#pragma unroll
            for (int nt = 0; nt < 2; ++nt) {
                v8f c = acc[mt][nt];
                c = __builtin_amdgcn_wmma_f32_16x16x32_bf16(
                        false, Ah[mt], false, Bh[nt], (short)0, c, false, false);
                c = __builtin_amdgcn_wmma_f32_16x16x32_bf16(
                        false, Ah[mt], false, Bl[nt], (short)0, c, false, false);
                c = __builtin_amdgcn_wmma_f32_16x16x32_bf16(
                        false, Al[mt], false, Bh[nt], (short)0, c, false, false);
                acc[mt][nt] = c;
            }

        // Convert+store next stage into the other buffer; the conversion VALU
        // co-executes with the XDL WMMAs above.  Race-free: other waves'
        // reads of buf^1 finished before this iteration's barrier.
        if (more)
            convert_store(nxt, smem + (pbuf ^ 1) * BUFHALF, ldRow, ldC);
        pbuf ^= 1;
    }

    // Epilogue: 16x16 f32 C/D layout — lane&15 = N, VGPR r = M row (+8 for
    // upper lane half).
    const int colBase = nBlock + wn * 32 + (lane & 15);
    const int rowBase = mBlock + wm * 64 + ((lane & 16) ? 8 : 0);
#pragma unroll
    for (int mt = 0; mt < 4; ++mt)
#pragma unroll
        for (int nt = 0; nt < 2; ++nt) {
            const int col = colBase + nt * 16;
#pragma unroll
            for (int r = 0; r < 8; ++r)
                Out[(size_t)(rowBase + mt * 16 + r) * N + col] = acc[mt][nt][r];
        }
}

// (M, K=in_features, N=out_features) matching GROUP_CFG (w is [N,K]).
static const int CFG[8][3] = {
    {4096, 1024, 2048},
    {8192, 2048, 1024},
    {2048,  512, 2048},
    {16384,1024, 1024},
    {1024, 2048, 2048},
    {4096,  768, 3072},
    {2048, 3072,  768},
    {8192, 1024, 4096},
};

extern "C" void kernel_launch(void* const* d_in, const int* in_sizes, int n_in,
                              void* d_out, int out_size, void* d_ws, size_t ws_size,
                              hipStream_t stream) {
    (void)in_sizes; (void)n_in; (void)out_size; (void)d_ws; (void)ws_size;
    float* out = (float*)d_out;
    size_t off = 0;
    for (int g = 0; g < 8; ++g) {
        const int M = CFG[g][0], K = CFG[g][1], N = CFG[g][2];
        // setup_inputs() inserts x{i}, w{i} interleaved.
        const float* x = (const float*)d_in[2 * g + 0];
        const float* w = (const float*)d_in[2 * g + 1];
        dim3 grid(N / BN, M / BM);
        group_gemm_bf16x3<<<grid, dim3(256), 0, stream>>>(x, w, out + off, M, N, K);
        off += (size_t)M * N;
    }
}
